// TemporalAssemblyGNN_2748779069972
// MI455X (gfx1250) — matrix operand order repaired and checked by reading, not compiled, not run on hardware
//
#include <hip/hip_runtime.h>
#include <hip/hip_bf16.h>

typedef __attribute__((ext_vector_type(16))) _Float16 v16h;
typedef __attribute__((ext_vector_type(8)))  _Float16 v8h;
typedef __attribute__((ext_vector_type(8)))  float    v8f;

namespace {
constexpr int B_  = 8, T_ = 16, NPG = 1000, EPG = 8000;
constexpr int G_  = B_ * T_;        // 128 graphs
constexpr int N_  = G_ * NPG;       // 128000 nodes
constexpr int E_  = G_ * EPG;       // 1024000 edges
constexpr int HD  = 128;            // node feature dim
constexpr int TD  = 256, TH = 8;    // transformer dim / heads
}

// ---------------- generic helpers ----------------

__global__ void fill_f32(float* p, float v, long n) {
  long i = (long)blockIdx.x * blockDim.x + threadIdx.x;
  if (i < n) p[i] = v;
}

__global__ void f32_to_f16(const float* __restrict__ a, _Float16* __restrict__ b, long n) {
  long i = (long)blockIdx.x * blockDim.x + threadIdx.x;
  if (i < n) b[i] = (_Float16)a[i];
}

__device__ __forceinline__ void atomicMaxF(float* addr, float val) {
  int iv = __float_as_int(val);
  if (iv >= 0) atomicMax((int*)addr, iv);
  else         atomicMin((unsigned int*)addr, (unsigned int)iv);
}

// generic dense: out[r,c] = act(A[r,:]@W[:,c] + bias[c] + resid[r,c])
// act: 0 = none, 1 = exact gelu
__global__ void dense_kernel(const float* __restrict__ A, const float* __restrict__ W,
                             const float* __restrict__ bias, const float* __restrict__ resid,
                             float* __restrict__ out, int M, int K, int C, int act) {
  long idx = (long)blockIdx.x * blockDim.x + threadIdx.x;
  if (idx >= (long)M * C) return;
  int c = (int)(idx % C);
  long r = idx / C;
  float s = bias ? bias[c] : 0.f;
  const float* a = A + r * K;
  for (int k = 0; k < K; ++k) s += a[k] * W[(long)k * C + c];
  if (resid) s += resid[idx];
  if (act == 1) s = 0.5f * s * (1.f + erff(s * 0.70710678118654752f));
  out[idx] = s;
}

// row layernorm over 256 features, one wave32 per row
__global__ void ln_rows256(float* __restrict__ dst, const float* __restrict__ src,
                           const float* __restrict__ g, const float* __restrict__ b, int rows) {
  int r    = (blockIdx.x * blockDim.x + threadIdx.x) >> 5;
  int lane = threadIdx.x & 31;
  if (r >= rows) return;
  const float* s = src + (long)r * 256;
  float v[8]; float s1 = 0.f, s2 = 0.f;
#pragma unroll
  for (int j = 0; j < 8; ++j) { v[j] = s[j * 32 + lane]; s1 += v[j]; s2 += v[j] * v[j]; }
#pragma unroll
  for (int o = 16; o > 0; o >>= 1) { s1 += __shfl_xor(s1, o); s2 += __shfl_xor(s2, o); }
  float m = s1 * (1.f / 256.f);
  float var = s2 * (1.f / 256.f) - m * m;
  float rr = rsqrtf(var + 1e-5f);
#pragma unroll
  for (int j = 0; j < 8; ++j) {
    int c = j * 32 + lane;
    dst[(long)r * 256 + c] = g[c] * (v[j] - m) * rr + b[c];
  }
}

// ---------------- WMMA GEMM: out[M x 128] = A[M x 128] @ W[128 x 128] ----------------
// A: f16 row-major. Wp: f16 packed in the CDNA5 B-operand layout:
//   Wp[((kb*8 + nb)*32 + lane)*16 + e]  holds  W[kb*32 + 8*(lane>>4) + (e&7) + 16*(e>>3), nb*16 + (lane&15)]
__global__ void pack_w128(const float* __restrict__ W, _Float16* __restrict__ Wp) {
  int t = blockIdx.x * blockDim.x + threadIdx.x;   // 4*8*32*16 = 16384
  if (t >= 16384) return;
  int e = t & 15, lane = (t >> 4) & 31, nb = (t >> 9) & 7, kb = t >> 12;
  int k = kb * 32 + 8 * (lane >> 4) + (e & 7) + 16 * (e >> 3);
  int n = nb * 16 + (lane & 15);
  Wp[t] = (_Float16)W[k * 128 + n];
}

// one wave per 16x16 C tile; block = 256 threads = 8 waves = full ntile row of one mtile
__global__ void wmma_gemm128(const _Float16* __restrict__ A, const _Float16* __restrict__ Wp,
                             float* __restrict__ out, const float* __restrict__ bias, int relu) {
  int mtile = blockIdx.x;
  int ntile = threadIdx.x >> 5;
  int lane  = threadIdx.x & 31;
  int rowA  = mtile * 16 + (lane & 15);
  int kg    = lane >> 4;
  float bv  = bias ? bias[ntile * 16 + (lane & 15)] : 0.f;
  v8f acc;
#pragma unroll
  for (int r = 0; r < 8; ++r) acc[r] = bv;
#pragma unroll
  for (int kb = 0; kb < 4; ++kb) {
    // A operand: lane&15 = row, lanes 0-15 hold K {0..7,16..23}, lanes 16-31 hold K {8..15,24..31}
    const _Float16* pa = A + (long)rowA * 128 + kb * 32 + kg * 8;
    v8h alo = *reinterpret_cast<const v8h*>(pa);
    v8h ahi = *reinterpret_cast<const v8h*>(pa + 16);
    v16h a;
#pragma unroll
    for (int e = 0; e < 8; ++e) { a[e] = alo[e]; a[e + 8] = ahi[e]; }
    v16h b = *reinterpret_cast<const v16h*>(Wp + (((kb * 8 + ntile) * 32) + lane) * 16);
    acc = __builtin_amdgcn_wmma_f32_16x16x32_f16(false, a, false, b, (short)0, acc, false, false);
  }
  // D layout: VGPR r -> row (r + 8*(lane>>4)), col = lane&15
  int col = ntile * 16 + (lane & 15);
  int row = mtile * 16 + 8 * (lane >> 4);
#pragma unroll
  for (int r = 0; r < 8; ++r) {
    float v = acc[r];
    if (relu) v = v > 0.f ? v : 0.f;
    out[(long)(row + r) * 128 + col] = v;
  }
}

// ---------------- GAT pieces ----------------

// deg[dst] += 1 ; easum[dst,0:12] += edge_attr[i,0:12]
__global__ void deg_easum_kernel(const int* __restrict__ dst, const float* __restrict__ ea,
                                 float* __restrict__ deg, float* __restrict__ easum) {
  int i = blockIdx.x * blockDim.x + threadIdx.x;
  if (i >= E_) return;
  int d = dst[i];
  atomicAdd(&deg[d], 1.f);
  const float* r = ea + (long)i * 12;
#pragma unroll
  for (int j = 0; j < 12; ++j) atomicAdd(&easum[d * 12 + j], r[j]);
}

// a_src[n,h] = xs[n,h*32:]·att_src[h]; a_dst likewise
__global__ void attdot_kernel(const float* __restrict__ xs, const float* __restrict__ att_src,
                              const float* __restrict__ att_dst,
                              float* __restrict__ a_src, float* __restrict__ a_dst) {
  long idx = (long)blockIdx.x * blockDim.x + threadIdx.x;   // n*4 + h
  if (idx >= (long)N_ * 4) return;
  int h = (int)(idx & 3);
  long n = idx >> 2;
  const float* p = xs + n * 128 + h * 32;
  float ss = 0.f, sd = 0.f;
#pragma unroll
  for (int c = 0; c < 32; ++c) { float v = p[c]; ss += v * att_src[h * 32 + c]; sd += v * att_dst[h * 32 + c]; }
  a_src[idx] = ss; a_dst[idx] = sd;
}

// Ve[k,h] = Σ_c We[k,h*32+c]·att_edge[h,c]; then M12[j,h] = W_enc[j,:]·Ve[:,h]; c[h]=b_enc·Ve[:,h]
// out: 48 floats of M12 (j*4+h) followed by 4 floats of cedge.  single block of 512 threads
__global__ void prep_edge_proj(const float* __restrict__ We, const float* __restrict__ att_edge,
                               const float* __restrict__ Wenc, const float* __restrict__ benc,
                               float* __restrict__ out) {
  __shared__ float Ve[512];
  int t = threadIdx.x;
  int k = t >> 2, h = t & 3;
  float s = 0.f;
  for (int c = 0; c < 32; ++c) s += We[k * 128 + h * 32 + c] * att_edge[h * 32 + c];
  Ve[k * 4 + h] = s;
  __syncthreads();
  if (t < 48) {
    int j = t >> 2, hh = t & 3;
    float a = 0.f;
    for (int k2 = 0; k2 < 128; ++k2) a += Wenc[j * 128 + k2] * Ve[k2 * 4 + hh];
    out[t] = a;
  } else if (t < 52) {
    int hh = t - 48;
    float a = 0.f;
    for (int k2 = 0; k2 < 128; ++k2) a += benc[k2] * Ve[k2 * 4 + hh];
    out[48 + hh] = a;
  }
}

// aeE[i,h] = edge_attr[i]·M12[:,h] + cedge[h]
__global__ void ae_edges_kernel(const float* __restrict__ ea, const float* __restrict__ M12c,
                                float* __restrict__ aeE) {
  __shared__ float M[52];
  if (threadIdx.x < 52) M[threadIdx.x] = M12c[threadIdx.x];
  __syncthreads();
  int i = blockIdx.x * blockDim.x + threadIdx.x;
  if (i >= E_) return;
  const float* r = ea + (long)i * 12;
  float f[12];
#pragma unroll
  for (int j = 0; j < 12; ++j) f[j] = r[j];
#pragma unroll
  for (int h = 0; h < 4; ++h) {
    float a = M[48 + h];
#pragma unroll
    for (int j = 0; j < 12; ++j) a += f[j] * M[j * 4 + h];
    aeE[(long)i * 4 + h] = a;
  }
}

// aeS[n,h] = (easum[n]·M12[:,h] + deg[n]*cedge[h]) / max(deg[n],1)
__global__ void ae_self_kernel(const float* __restrict__ easum, const float* __restrict__ deg,
                               const float* __restrict__ M12c, float* __restrict__ aeS) {
  __shared__ float M[52];
  if (threadIdx.x < 52) M[threadIdx.x] = M12c[threadIdx.x];
  __syncthreads();
  int n = blockIdx.x * blockDim.x + threadIdx.x;
  if (n >= N_) return;
  float dg = deg[n];
  float inv = 1.f / fmaxf(dg, 1.f);
  const float* r = easum + (long)n * 12;
#pragma unroll
  for (int h = 0; h < 4; ++h) {
    float a = dg * M[48 + h];
#pragma unroll
    for (int j = 0; j < 12; ++j) a += r[j] * M[j * 4 + h];
    aeS[(long)n * 4 + h] = a * inv;
  }
}

__device__ __forceinline__ float lrelu02(float x) { return x > 0.f ? x : 0.2f * x; }

__global__ void pass1_edges(const int* __restrict__ src, const int* __restrict__ dst,
                            const float* __restrict__ a_src, const float* __restrict__ a_dst,
                            const float* __restrict__ aeE, float* __restrict__ mx) {
  int i = blockIdx.x * blockDim.x + threadIdx.x;
  if (i >= E_) return;
  int s = src[i], d = dst[i];
#pragma unroll
  for (int h = 0; h < 4; ++h) {
    float l = lrelu02(a_src[s * 4 + h] + a_dst[d * 4 + h] + aeE[(long)i * 4 + h]);
    atomicMaxF(&mx[d * 4 + h], l);
  }
}

__global__ void pass1_self(const float* __restrict__ a_src, const float* __restrict__ a_dst,
                           const float* __restrict__ aeS, float* __restrict__ mx) {
  int n = blockIdx.x * blockDim.x + threadIdx.x;
  if (n >= N_) return;
#pragma unroll
  for (int h = 0; h < 4; ++h) {
    float l = lrelu02(a_src[n * 4 + h] + a_dst[n * 4 + h] + aeS[(long)n * 4 + h]);
    atomicMaxF(&mx[n * 4 + h], l);
  }
}

// wave per edge: ex = exp(logit - mx[dst]); denom[dst] += ex; agg[dst,:] += ex * xs[src,:]
__global__ void pass2_edges(const int* __restrict__ src, const int* __restrict__ dst,
                            const float* __restrict__ a_src, const float* __restrict__ a_dst,
                            const float* __restrict__ aeE, const float* __restrict__ mx,
                            const float* __restrict__ xs, float* __restrict__ agg,
                            float* __restrict__ denom) {
  int wid  = (blockIdx.x * blockDim.x + threadIdx.x) >> 5;
  int lane = threadIdx.x & 31;
  if (wid >= E_) return;
  int s = src[wid], d = dst[wid];
  float ex[4];
#pragma unroll
  for (int h = 0; h < 4; ++h) {
    float l = lrelu02(a_src[s * 4 + h] + a_dst[d * 4 + h] + aeE[(long)wid * 4 + h]);
    ex[h] = __expf(l - mx[d * 4 + h]);
    if (lane == 0) atomicAdd(&denom[d * 4 + h], ex[h]);
  }
#pragma unroll
  for (int h = 0; h < 4; ++h) {
    int c = h * 32 + lane;
    atomicAdd(&agg[(long)d * 128 + c], ex[h] * xs[(long)s * 128 + c]);
  }
}

__global__ void pass2_self(const float* __restrict__ a_src, const float* __restrict__ a_dst,
                           const float* __restrict__ aeS, const float* __restrict__ mx,
                           const float* __restrict__ xs, float* __restrict__ agg,
                           float* __restrict__ denom) {
  int n    = (blockIdx.x * blockDim.x + threadIdx.x) >> 5;
  int lane = threadIdx.x & 31;
  if (n >= N_) return;
  float ex[4];
#pragma unroll
  for (int h = 0; h < 4; ++h) {
    float l = lrelu02(a_src[n * 4 + h] + a_dst[n * 4 + h] + aeS[(long)n * 4 + h]);
    ex[h] = __expf(l - mx[n * 4 + h]);
    if (lane == 0) atomicAdd(&denom[n * 4 + h], ex[h]);
  }
#pragma unroll
  for (int h = 0; h < 4; ++h) {
    int c = h * 32 + lane;
    atomicAdd(&agg[(long)n * 128 + c], ex[h] * xs[(long)n * 128 + c]);
  }
}

// h[n,:] += relu(LN(agg[n,:]/denom[n,head] + bias))  (wave per node, 4 channels/lane)
__global__ void gat_finalize(float* __restrict__ h, const float* __restrict__ agg,
                             const float* __restrict__ denom, const float* __restrict__ bias,
                             const float* __restrict__ ln_g, const float* __restrict__ ln_b) {
  int n    = (blockIdx.x * blockDim.x + threadIdx.x) >> 5;
  int lane = threadIdx.x & 31;
  if (n >= N_) return;
  float v[4]; float s1 = 0.f, s2 = 0.f;
#pragma unroll
  for (int j = 0; j < 4; ++j) {
    int c = j * 32 + lane;                       // head = j
    float val = agg[(long)n * 128 + c] / denom[n * 4 + j] + bias[c];
    v[j] = val; s1 += val; s2 += val * val;
  }
#pragma unroll
  for (int o = 16; o > 0; o >>= 1) { s1 += __shfl_xor(s1, o); s2 += __shfl_xor(s2, o); }
  float m = s1 * (1.f / 128.f);
  float var = s2 * (1.f / 128.f) - m * m;
  float r = rsqrtf(var + 1e-5f);
#pragma unroll
  for (int j = 0; j < 4; ++j) {
    int c = j * 32 + lane;
    float y = ln_g[c] * (v[j] - m) * r + ln_b[c];
    h[(long)n * 128 + c] += y > 0.f ? y : 0.f;
  }
}

// ---------------- gating / pooling ----------------

__global__ void gate2_kernel(const float* __restrict__ g1, const float* __restrict__ w,
                             const float* __restrict__ b, float* __restrict__ gate) {
  int n    = (blockIdx.x * blockDim.x + threadIdx.x) >> 5;
  int lane = threadIdx.x & 31;
  if (n >= N_) return;
  const float* r = g1 + (long)n * 128;
  float s = 0.f;
#pragma unroll
  for (int j = 0; j < 4; ++j) s += r[j * 32 + lane] * w[j * 32 + lane];
#pragma unroll
  for (int o = 16; o > 0; o >>= 1) s += __shfl_xor(s, o);
  if (lane == 0) gate[n] = s + b[0];
}

// one block per graph: softmax over its 1000 nodes, weighted feature sum
__global__ void graph_pool(const float* __restrict__ h, const float* __restrict__ gate,
                           float* __restrict__ graphv) {
  __shared__ float sred[256];
  __shared__ float wbuf[NPG];
  int g = blockIdx.x, t = threadIdx.x;
  const float* gg = gate + (long)g * NPG;
  float mx = -1e30f;
  for (int i = t; i < NPG; i += 256) mx = fmaxf(mx, gg[i]);
  sred[t] = mx; __syncthreads();
  for (int s = 128; s > 0; s >>= 1) { if (t < s) sred[t] = fmaxf(sred[t], sred[t + s]); __syncthreads(); }
  float m = sred[0]; __syncthreads();
  float sum = 0.f;
  for (int i = t; i < NPG; i += 256) { float e = __expf(gg[i] - m); wbuf[i] = e; sum += e; }
  sred[t] = sum; __syncthreads();
  for (int s = 128; s > 0; s >>= 1) { if (t < s) sred[t] += sred[t + s]; __syncthreads(); }
  float inv = 1.f / sred[0]; __syncthreads();
  if (t < 128) {
    const float* hb = h + (long)g * NPG * 128;
    float acc = 0.f;
    for (int i = 0; i < NPG; ++i) acc += wbuf[i] * hb[(long)i * 128 + t];
    graphv[g * 128 + t] = acc * inv;
  }
}

// hs[r,o] = concat(graph[r], topo[r]) @ Winp[:,o] + b[o] + pos[t,o]
__global__ void in_proj_kernel(const float* __restrict__ graphv, const float* __restrict__ topo,
                               const float* __restrict__ W, const float* __restrict__ b,
                               const float* __restrict__ pos, float* __restrict__ hs) {
  int idx = blockIdx.x * blockDim.x + threadIdx.x;   // r*256 + o
  if (idx >= G_ * TD) return;
  int o = idx & 255, r = idx >> 8, t = r & 15;
  float s = b[o] + pos[t * TD + o];
  const float* gv = graphv + r * 128;
  for (int k = 0; k < 128; ++k) s += gv[k] * W[k * TD + o];
  const float* tp = topo + r * 7;
#pragma unroll
  for (int k = 0; k < 7; ++k) s += tp[k] * W[(128 + k) * TD + o];
  hs[idx] = s;
}

// ---------------- transformer ----------------

// thread per (b, head, t): full softmax attention over T=16
__global__ void attn_kernel(const float* __restrict__ qkv, const unsigned char* __restrict__ mask,
                            float* __restrict__ obuf) {
  int idx = blockIdx.x * blockDim.x + threadIdx.x;   // b*128 + h*16 + t
  if (idx >= B_ * TH * T_) return;
  int t = idx & 15, h = (idx >> 4) & 7, b = idx >> 7;
  const float* q = qkv + (long)(b * T_ + t) * 768 + h * 32;
  float sc[16]; float mxv = -1e30f;
  for (int s = 0; s < T_; ++s) {
    const float* k = qkv + (long)(b * T_ + s) * 768 + 256 + h * 32;
    float d = 0.f;
#pragma unroll
    for (int j = 0; j < 32; ++j) d += q[j] * k[j];
    d *= 0.17677669529663687f;                       // 1/sqrt(32)
    if (!mask[b * T_ + s]) d = -1e9f;
    sc[s] = d; mxv = fmaxf(mxv, d);
  }
  float sum = 0.f;
#pragma unroll
  for (int s = 0; s < 16; ++s) { sc[s] = __expf(sc[s] - mxv); sum += sc[s]; }
  float inv = 1.f / sum;
  float o[32];
#pragma unroll
  for (int j = 0; j < 32; ++j) o[j] = 0.f;
  for (int s = 0; s < T_; ++s) {
    const float* v = qkv + (long)(b * T_ + s) * 768 + 512 + h * 32;
    float w = sc[s] * inv;
#pragma unroll
    for (int j = 0; j < 32; ++j) o[j] += w * v[j];
  }
  float* op = obuf + (long)(b * T_ + t) * 256 + h * 32;
#pragma unroll
  for (int j = 0; j < 32; ++j) op[j] = o[j];
}

__global__ void pool_kernel(const float* __restrict__ hs, const unsigned char* __restrict__ mask,
                            float* __restrict__ pooled) {
  int idx = blockIdx.x * blockDim.x + threadIdx.x;   // b*256 + c
  if (idx >= B_ * TD) return;
  int c = idx & 255, b = idx >> 8;
  float s = 0.f, wsum = 0.f;
  for (int t = 0; t < T_; ++t) {
    float m = mask[b * T_ + t] ? 1.f : 0.f;
    s += m * hs[(long)(b * T_ + t) * 256 + c];
    wsum += m;
  }
  pooled[idx] = s / fmaxf(wsum, 1.f);
}

// ---------------- launch ----------------

extern "C" void kernel_launch(void* const* d_in, const int* in_sizes, int n_in,
                              void* d_out, int out_size, void* d_ws, size_t ws_size,
                              hipStream_t stream) {
  (void)in_sizes; (void)n_in; (void)out_size; (void)ws_size;
  auto F = [&](int i) { return (const float*)d_in[i]; };
  // data inputs (setup_inputs insertion order)
  const float* x        = F(0);
  const int*   eidx     = (const int*)d_in[1];
  const float* ea       = F(2);
  const float* topo     = F(4);
  const unsigned char* mask = (const unsigned char*)d_in[5];
  const int* src = eidx;
  const int* dst = eidx + E_;
  // params: assumed flattened in recursive dict-insertion order of setup_inputs()
  const int P0 = 6;
  const float* neW = F(P0 + 0); const float* neB = F(P0 + 1);
  const float* eeW = F(P0 + 2); const float* eeB = F(P0 + 3);
  const int GAT0 = P0 + 4;                 // gat[l]: W,att_src,att_dst,We,att_edge,b,ln_g,ln_b
  const int GP   = GAT0 + 24;              // gate1.W,gate1.b,gate2.W,gate2.b,in_proj.W,in_proj.b,pos
  const int TF0  = GP + 7;                 // tf[l]: Wqkv,bqkv,Wo,bo,ln1_g,ln1_b,ln2_g,ln2_b,W1,b1,W2,b2
  const int FIN  = TF0 + 24;               // final.W, final.b

  // workspace carve (256B aligned)
  size_t off = 0;
  auto alloc = [&](size_t bytes) { void* p = (char*)d_ws + off; off += (bytes + 255) & ~(size_t)255; return p; };
  float* h      = (float*)alloc((size_t)N_ * 128 * 4);
  float* xs     = (float*)alloc((size_t)N_ * 128 * 4);
  float* agg    = (float*)alloc((size_t)N_ * 128 * 4);
  _Float16* hh  = (_Float16*)agg;          // f16 staging aliased over agg (agg zeroed after GEMM)
  _Float16* Wp  = (_Float16*)alloc(16384 * 2);
  float* deg    = (float*)alloc((size_t)N_ * 4);
  float* easum  = (float*)alloc((size_t)N_ * 12 * 4);
  float* a_src  = (float*)alloc((size_t)N_ * 4 * 4);
  float* a_dst  = (float*)alloc((size_t)N_ * 4 * 4);
  float* aeE    = (float*)alloc((size_t)E_ * 4 * 4);
  float* aeS    = (float*)alloc((size_t)N_ * 4 * 4);
  float* mx     = (float*)alloc((size_t)N_ * 4 * 4);
  float* denom  = (float*)alloc((size_t)N_ * 4 * 4);
  float* gatev  = (float*)alloc((size_t)N_ * 4);
  float* M12c   = (float*)alloc(256);
  float* graphv = (float*)alloc((size_t)G_ * 128 * 4);
  float* hs     = (float*)alloc((size_t)G_ * TD * 4);
  float* qkv    = (float*)alloc((size_t)G_ * 3 * TD * 4);
  float* obuf   = (float*)alloc((size_t)G_ * TD * 4);
  float* tmpb   = (float*)alloc((size_t)G_ * TD * 4);
  float* f1     = (float*)alloc((size_t)G_ * 4 * TD * 4);
  float* pooled = (float*)alloc((size_t)B_ * TD * 4);

  const int TB = 256;
  auto blk = [&](long n) { return (int)((n + TB - 1) / TB); };

  // 1. node encoder: h = x @ neW + neB   (N x 42 @ 42 x 128)
  dense_kernel<<<blk((long)N_ * 128), TB, 0, stream>>>(x, neW, neB, nullptr, h, N_, 42, 128, 0);

  // 2. degree + edge-attr segment sums (folded e_self precompute)
  fill_f32<<<blk(N_), TB, 0, stream>>>(deg, 0.f, N_);
  fill_f32<<<blk((long)N_ * 12), TB, 0, stream>>>(easum, 0.f, (long)N_ * 12);
  deg_easum_kernel<<<blk(E_), TB, 0, stream>>>(dst, ea, deg, easum);

  // 3. three GAT layers
  for (int l = 0; l < 3; ++l) {
    const float* W  = F(GAT0 + l * 8 + 0);
    const float* as = F(GAT0 + l * 8 + 1);
    const float* ad = F(GAT0 + l * 8 + 2);
    const float* We = F(GAT0 + l * 8 + 3);
    const float* ae_ = F(GAT0 + l * 8 + 4);
    const float* bb = F(GAT0 + l * 8 + 5);
    const float* lg = F(GAT0 + l * 8 + 6);
    const float* lb = F(GAT0 + l * 8 + 7);

    f32_to_f16<<<blk((long)N_ * 128), TB, 0, stream>>>(h, hh, (long)N_ * 128);
    pack_w128<<<64, TB, 0, stream>>>(W, Wp);
    wmma_gemm128<<<N_ / 16, TB, 0, stream>>>(hh, Wp, xs, nullptr, 0);

    attdot_kernel<<<blk((long)N_ * 4), TB, 0, stream>>>(xs, as, ad, a_src, a_dst);
    prep_edge_proj<<<1, 512, 0, stream>>>(We, ae_, eeW, eeB, M12c);
    ae_edges_kernel<<<blk(E_), TB, 0, stream>>>(ea, M12c, aeE);
    ae_self_kernel<<<blk(N_), TB, 0, stream>>>(easum, deg, M12c, aeS);

    fill_f32<<<blk((long)N_ * 4), TB, 0, stream>>>(mx, -1e30f, (long)N_ * 4);
    fill_f32<<<blk((long)N_ * 4), TB, 0, stream>>>(denom, 0.f, (long)N_ * 4);
    fill_f32<<<blk((long)N_ * 128), TB, 0, stream>>>(agg, 0.f, (long)N_ * 128);

    pass1_edges<<<blk(E_), TB, 0, stream>>>(src, dst, a_src, a_dst, aeE, mx);
    pass1_self<<<blk(N_), TB, 0, stream>>>(a_src, a_dst, aeS, mx);
    pass2_edges<<<blk((long)E_ * 32), TB, 0, stream>>>(src, dst, a_src, a_dst, aeE, mx, xs, agg, denom);
    pass2_self<<<blk((long)N_ * 32), TB, 0, stream>>>(a_src, a_dst, aeS, mx, xs, agg, denom);
    gat_finalize<<<blk((long)N_ * 32), TB, 0, stream>>>(h, agg, denom, bb, lg, lb);
  }

  // 4. gating + graph pooling
  f32_to_f16<<<blk((long)N_ * 128), TB, 0, stream>>>(h, hh, (long)N_ * 128);
  pack_w128<<<64, TB, 0, stream>>>(F(GP + 0), Wp);
  wmma_gemm128<<<N_ / 16, TB, 0, stream>>>(hh, Wp, xs, F(GP + 1), 1);   // relu(h@gate1+b)
  gate2_kernel<<<blk((long)N_ * 32), TB, 0, stream>>>(xs, F(GP + 2), F(GP + 3), gatev);
  graph_pool<<<G_, TB, 0, stream>>>(h, gatev, graphv);

  // 5. in_proj + pos
  in_proj_kernel<<<blk((long)G_ * TD), TB, 0, stream>>>(graphv, topo, F(GP + 4), F(GP + 5), F(GP + 6), hs);

  // 6. transformer layers
  for (int l = 0; l < 2; ++l) {
    const int tb = TF0 + l * 12;
    dense_kernel<<<blk((long)G_ * 768), TB, 0, stream>>>(hs, F(tb + 0), F(tb + 1), nullptr, qkv, G_, 256, 768, 0);
    attn_kernel<<<blk(B_ * TH * T_), TB, 0, stream>>>(qkv, mask, obuf);
    dense_kernel<<<blk((long)G_ * 256), TB, 0, stream>>>(obuf, F(tb + 2), F(tb + 3), hs, tmpb, G_, 256, 256, 0);
    ln_rows256<<<blk((long)G_ * 32), TB, 0, stream>>>(hs, tmpb, F(tb + 4), F(tb + 5), G_);
    dense_kernel<<<blk((long)G_ * 1024), TB, 0, stream>>>(hs, F(tb + 8), F(tb + 9), nullptr, f1, G_, 256, 1024, 1);
    dense_kernel<<<blk((long)G_ * 256), TB, 0, stream>>>(f1, F(tb + 10), F(tb + 11), hs, tmpb, G_, 1024, 256, 0);
    ln_rows256<<<blk((long)G_ * 32), TB, 0, stream>>>(hs, tmpb, F(tb + 6), F(tb + 7), G_);
  }

  // 7. masked mean pool + final projection
  pool_kernel<<<blk((long)B_ * TD), TB, 0, stream>>>(hs, mask, pooled);
  dense_kernel<<<blk((long)B_ * 256), TB, 0, stream>>>(pooled, F(FIN + 0), F(FIN + 1), nullptr,
                                                       (float*)d_out, B_, 256, 256, 0);
}
